// LogStorageDiagonalCell_62285615727031
// MI455X (gfx1250) — matrix operand order repaired and checked by reading, not compiled, not run on hardware
//
#include <hip/hip_runtime.h>

typedef __bf16 bf16_t;
typedef __attribute__((ext_vector_type(16))) __bf16 v16bf;
typedef __attribute__((ext_vector_type(8)))  __bf16 v8bf;
typedef __attribute__((ext_vector_type(8)))  float  v8f;

constexpr int kT = 256;
constexpr int kB = 16;
constexpr int kD = 1024;
constexpr int kNWG = 64;                        // persistent workgroups
constexpr size_t kTBD = (size_t)kT * kB * kD;   // 4194304
constexpr size_t kBD  = (size_t)kB * kD;        // 16384
constexpr size_t kDD  = (size_t)kD * kD;        // 1048576

#if __has_builtin(__builtin_amdgcn_global_load_async_to_lds_b128) && \
    __has_builtin(__builtin_amdgcn_s_wait_asynccnt)
#define HAVE_ASYNC_LDS 1
typedef int v4i __attribute__((__vector_size__(16)));
typedef __attribute__((address_space(1))) v4i gv4i;   // global int4
typedef __attribute__((address_space(3))) v4i lv4i;   // LDS int4
#else
#define HAVE_ASYNC_LDS 0
#endif

static __device__ __forceinline__ v16bf cat8(v8bf a, v8bf b) {
  return __builtin_shufflevector(a, b, 0,1,2,3,4,5,6,7,8,9,10,11,12,13,14,15);
}

// ---------------------------------------------------------------------------
// Kernel 0: f32 -> bf16 conversion of x, W_x, W_delta, W_out; zero barrier slots
// ---------------------------------------------------------------------------
__global__ void prep_kernel(const float* __restrict__ x,
                            const float* __restrict__ Wx,
                            const float* __restrict__ Wd,
                            const float* __restrict__ Wo,
                            bf16_t* __restrict__ xb,
                            bf16_t* __restrict__ Wxb,
                            bf16_t* __restrict__ Wdb,
                            bf16_t* __restrict__ Wob,
                            unsigned* __restrict__ bar) {
  size_t i = (size_t)blockIdx.x * blockDim.x + threadIdx.x;
  size_t stride = (size_t)gridDim.x * blockDim.x;
  for (size_t j = i; j < kTBD; j += stride) xb[j] = (bf16_t)x[j];
  for (size_t j = i; j < kDD; j += stride) {
    Wxb[j] = (bf16_t)Wx[j];
    Wdb[j] = (bf16_t)Wd[j];
    Wob[j] = (bf16_t)Wo[j];
  }
  if (i < (size_t)kT) bar[i] = 0u;
}

// ---------------------------------------------------------------------------
// Kernel 1: xw = x @ W^T + bias  (z=0: W_x/b, z=1: W_delta/b_delta)
// M=4096, N=1024, K=1024. WG = 8 waves; WG tile 64(M) x 256(N);
// wave tile 32x64 = 2x4 WMMA tiles (8 WMMA per 12 b128 loads per k-step).
// ---------------------------------------------------------------------------
__global__ __launch_bounds__(256, 1) void proj_gemm_kernel(
    const bf16_t* __restrict__ xb,
    const bf16_t* __restrict__ Wxb,
    const bf16_t* __restrict__ Wdb,
    const float* __restrict__ bias_x,
    const float* __restrict__ bias_d,
    float* __restrict__ xw, float* __restrict__ xd) {
  const bf16_t* W   = (blockIdx.z == 0) ? Wxb : Wdb;
  const float* bias = (blockIdx.z == 0) ? bias_x : bias_d;
  float* out        = (blockIdx.z == 0) ? xw : xd;

  const int wave = threadIdx.x >> 5;
  const int lane = threadIdx.x & 31;
  const int lhalf = lane >> 4;   // 0/1
  const int lmod  = lane & 15;
  const int vm = wave & 1, vn = wave >> 1;     // 2(m) x 4(n) wave grid
  const int m0 = blockIdx.y * 64 + vm * 32;
  const int n0 = blockIdx.x * 256 + vn * 64;

  v8f acc[2][4] = {};

  for (int k0 = 0; k0 < kD; k0 += 32) {
    v16bf a[2], bm[4];
    // A fragment (16x32 bf16): lane<16: M=lane, K={0..7,16..23}; lane>=16: K={8..15,24..31}
    #pragma unroll
    for (int mi = 0; mi < 2; ++mi) {
      const bf16_t* p = xb + (size_t)(m0 + mi * 16 + lmod) * kD + k0 + lhalf * 8;
      a[mi] = cat8(*(const v8bf*)p, *(const v8bf*)(p + 16));
    }
    // B fragment (32x16 bf16): lane<16: N=lane, K={0..7,16..23}; lane>=16: K={8..15,24..31}
    #pragma unroll
    for (int ni = 0; ni < 4; ++ni) {
      const bf16_t* p = W + (size_t)(n0 + ni * 16 + lmod) * kD + k0 + lhalf * 8;
      bm[ni] = cat8(*(const v8bf*)p, *(const v8bf*)(p + 16));
    }
    #pragma unroll
    for (int mi = 0; mi < 2; ++mi)
      #pragma unroll
      for (int ni = 0; ni < 4; ++ni)
        acc[mi][ni] = __builtin_amdgcn_wmma_f32_16x16x32_bf16(
            false, a[mi], false, bm[ni], (short)0, acc[mi][ni], false, false);
  }
  #pragma unroll
  for (int mi = 0; mi < 2; ++mi)
    #pragma unroll
    for (int ni = 0; ni < 4; ++ni) {
      const int n = n0 + ni * 16 + lmod;
      const float bv = bias[n];
      #pragma unroll
      for (int r = 0; r < 8; ++r) {
        const int m = m0 + mi * 16 + r + lhalf * 8;
        out[(size_t)m * kD + n] = acc[mi][ni][r] + bv;
      }
    }
}

// ---------------------------------------------------------------------------
// Kernel 2: persistent pipelined scan. 64 WGs x 256 threads (= B*D state elems).
// Pipeline: pass barrier(t) -> async-stage h(t)->LDS -> Phase A(t+1) ->
// arrive barrier(t+1) -> Phase B(t). Barrier latency hidden behind WMMA work.
// Triple-buffered h/compete traffic through L2.
// ---------------------------------------------------------------------------
__global__ __launch_bounds__(256, 1) void scan_kernel(
    const float* __restrict__ xw, const float* __restrict__ xd,
    const float* __restrict__ log_h0, const float* __restrict__ sign_h0,
    const float* __restrict__ r_h, const bf16_t* __restrict__ Wob,
    float* __restrict__ out_output, float* __restrict__ out_logh,
    float* __restrict__ out_signh, float* __restrict__ out_hlin,
    bf16_t* __restrict__ hbuf,   // [3][B*D] bf16
    float* __restrict__ cbuf,    // [3][B*D] f32
    unsigned* __restrict__ bar)  // [T]
{
  __shared__ __align__(16) bf16_t lWout[16 * kD];   // 32 KB: W_out rows e0..e0+15
  __shared__ __align__(16) bf16_t lh[kB * kD];      // 32 KB: staged h matrix
  __shared__ __align__(16) float  lred[8 * 256];    // 8 KB: per-wave partial tiles

  const int wg   = blockIdx.x;        // 0..63
  const int tid  = threadIdx.x;
  const int wave = tid >> 5;
  const int lane = tid & 31;
  const int gw   = wg * 8 + wave;     // 0..511 global wave id
  const int bidx = gw >> 5;           // batch 0..15
  const int grp  = gw & 31;           // softmax group 0..31
  const int d    = grp * 32 + lane;   // owned hidden index
  const int e0   = wg * 16;           // owned output-column tile
  const int lhalf = lane >> 4, lmod = lane & 15;

  for (int i = tid; i < 16 * kD; i += 256)
    lWout[i] = Wob[(size_t)(e0 + (i >> 10)) * kD + (i & (kD - 1))];
  __syncthreads();

  const size_t sidx = (size_t)bidx * kD + d;
  const float l0 = log_h0[sidx], s0 = sign_h0[sidx];
  float h = s0 * expf(l0);
  out_logh[sidx]  = l0;
  out_signh[sidx] = s0;
  const float rh = r_h[d];

  auto phaseA = [&](int tt, float xwv, float xdv) {
    const size_t idx = (size_t)tt * kBD + sidx;
    const float v     = xwv + rh * h;
    const float cand  = tanhf(v);
    const float delta = 1.f / (1.f + expf(-xdv));
    h = (1.f - delta) * h + delta * cand;
    out_hlin[idx] = h;
    out_logh[idx + kBD]  = logf(fmaxf(fabsf(h), 1e-38f));
    out_signh[idx + kBD] = (h >= 0.f) ? 1.f : -1.f;
    float mx = h;
    #pragma unroll
    for (int o = 16; o > 0; o >>= 1) mx = fmaxf(mx, __shfl_xor(mx, o));
    const float ex = expf(h - mx);
    float sm = ex;
    #pragma unroll
    for (int o = 16; o > 0; o >>= 1) sm += __shfl_xor(sm, o);
    const int bb = tt % 3;
    hbuf[(size_t)bb * kBD + sidx] = (bf16_t)h;
    cbuf[(size_t)bb * kBD + sidx] = ex / sm;
  };

  // Prologue: step 0 recurrence, arrive at barrier 0, prefetch step 1 inputs.
  phaseA(0, xw[sidx], xd[sidx]);
  __threadfence();
  __syncthreads();
  if (tid == 0)
    __hip_atomic_fetch_add(&bar[0], 1u, __ATOMIC_RELEASE, __HIP_MEMORY_SCOPE_AGENT);
  float xwn = xw[sidx + kBD];
  float xdn = xd[sidx + kBD];

  for (int t = 0; t < kT; ++t) {
    const int buf = t % 3;
    // ---- pass barrier(t): h(t)/compete(t) globally visible -----------------
    if (tid == 0) {
      while (__hip_atomic_load(&bar[t], __ATOMIC_ACQUIRE, __HIP_MEMORY_SCOPE_AGENT) < (unsigned)kNWG)
        __builtin_amdgcn_s_sleep(1);
    }
    __syncthreads();
    __builtin_amdgcn_fence(__ATOMIC_ACQUIRE, "agent");

#if HAVE_ASYNC_LDS
    // ---- fire-and-forget: stage h(t) [32KB] into LDS (ASYNCcnt path) -------
    {
      const bf16_t* src = hbuf + (size_t)buf * kBD;
      #pragma unroll
      for (int i = 0; i < 8; ++i) {
        const int off = (i * 256 + tid) * 8;   // 8 bf16 = 16B per thread-chunk
        __builtin_amdgcn_global_load_async_to_lds_b128(
            (gv4i*)(uintptr_t)(src + off),
            (lv4i*)(unsigned)(uintptr_t)(&lh[off]), 0, 0);
      }
    }
#endif
    // ---- Phase A(t+1): overlapped with async staging / barrier traffic -----
    if (t + 1 < kT) {
      phaseA(t + 1, xwn, xdn);
      const int tp = (t + 2 < kT) ? (t + 2) : (kT - 1);
      xwn = xw[(size_t)tp * kBD + sidx];
      xdn = xd[(size_t)tp * kBD + sidx];
      __threadfence();
    }
#if HAVE_ASYNC_LDS
    __builtin_amdgcn_s_wait_asynccnt(0);
#endif
    __syncthreads();
    if (t + 1 < kT && tid == 0)
      __hip_atomic_fetch_add(&bar[t + 1], 1u, __ATOMIC_RELEASE, __HIP_MEMORY_SCOPE_AGENT);

    // ---- Phase B(t): wh[:, e0:e0+16] = h(t) @ W_out_sliceT -----------------
#if HAVE_ASYNC_LDS
    const bf16_t* hA = lh;                        // ds_load path
#else
    const bf16_t* hA = hbuf + (size_t)buf * kBD;  // global path (triple-buffered)
#endif
    v8f acc = {};
    #pragma unroll
    for (int j = 0; j < 4; ++j) {
      const int k0 = wave * 128 + j * 32;
      const bf16_t* pa = hA + (size_t)lmod * kD + k0 + lhalf * 8;
      v16bf af = cat8(*(const v8bf*)pa, *(const v8bf*)(pa + 16));
      const bf16_t* pb = &lWout[lmod * kD + k0 + lhalf * 8];
      v16bf bfr = cat8(*(const v8bf*)pb, *(const v8bf*)(pb + 16));
      acc = __builtin_amdgcn_wmma_f32_16x16x32_bf16(false, af, false, bfr,
                                                    (short)0, acc, false, false);
    }
    #pragma unroll
    for (int r = 0; r < 8; ++r) lred[wave * 256 + r * 32 + lane] = acc[r];
    __syncthreads();
    {
      float s = 0.f;
      #pragma unroll
      for (int vv = 0; vv < 8; ++vv) s += lred[vv * 256 + tid];
      const int r = tid >> 5, ln = tid & 31;
      const int m = r + ((ln >> 4) << 3);          // batch row
      const int n = e0 + (ln & 15);                // output column
      const float silu = s / (1.f + expf(-s));
      const float comp = cbuf[(size_t)buf * kBD + (size_t)m * kD + n];
      out_output[(size_t)t * kBD + (size_t)m * kD + n] = comp * silu;
    }
    __syncthreads();  // lred / lh reused next step
  }
}

// ---------------------------------------------------------------------------
extern "C" void kernel_launch(void* const* d_in, const int* in_sizes, int n_in,
                              void* d_out, int out_size, void* d_ws, size_t ws_size,
                              hipStream_t stream) {
  const float* x       = (const float*)d_in[0];
  const float* log_h0  = (const float*)d_in[1];
  const float* sign_h0 = (const float*)d_in[2];
  const float* W_x     = (const float*)d_in[3];
  const float* r_h     = (const float*)d_in[4];
  const float* W_delta = (const float*)d_in[5];
  const float* W_out   = (const float*)d_in[6];
  const float* b       = (const float*)d_in[7];
  const float* b_delta = (const float*)d_in[8];
  (void)in_sizes; (void)n_in; (void)out_size; (void)ws_size;

  // Output layout: output[T,B,D] | log_h[T+1,B,D] | sign_h[T+1,B,D] | h_linear[T,B,D]
  float* out_output = (float*)d_out;
  float* out_logh   = out_output + kTBD;
  float* out_signh  = out_logh + (size_t)(kT + 1) * kBD;
  float* out_hlin   = out_signh + (size_t)(kT + 1) * kBD;

  char* ws = (char*)d_ws;
  size_t off = 0;
  auto carve = [&](size_t bytes) { char* p = ws + off; off = (off + bytes + 255) & ~(size_t)255; return p; };
  bf16_t* xb   = (bf16_t*)carve(kTBD * 2);
  bf16_t* Wxb  = (bf16_t*)carve(kDD * 2);
  bf16_t* Wdb  = (bf16_t*)carve(kDD * 2);
  bf16_t* Wob  = (bf16_t*)carve(kDD * 2);
  float*  xw   = (float*)carve(kTBD * 4);
  float*  xd   = (float*)carve(kTBD * 4);
  bf16_t* hbuf = (bf16_t*)carve(3 * kBD * 2);
  float*  cbuf = (float*)carve(3 * kBD * 4);
  unsigned* bar = (unsigned*)carve(kT * 4);

  prep_kernel<<<4096, 256, 0, stream>>>(x, W_x, W_delta, W_out, xb, Wxb, Wdb, Wob, bar);

  dim3 ggrid(kD / 256, (kT * kB) / 64, 2);
  proj_gemm_kernel<<<ggrid, 256, 0, stream>>>(xb, Wxb, Wdb, b, b_delta, xw, xd);

  scan_kernel<<<kNWG, 256, 0, stream>>>(xw, xd, log_h0, sign_h0, r_h, Wob,
                                        out_output, out_logh, out_signh, out_hlin,
                                        hbuf, cbuf, bar);
}
